// Decoder_81415400063199
// MI455X (gfx1250) — matrix operand order repaired and checked by reading, not compiled, not run on hardware
//
#include <hip/hip_runtime.h>

typedef unsigned short ushort_t;
typedef unsigned int uint_t;

typedef __bf16 bf16_t;
typedef bf16_t v16bf __attribute__((ext_vector_type(16)));
typedef float  v8f   __attribute__((ext_vector_type(8)));

#define ENC 512
#define DEC 256
#define NH  8
#define HD  64
#define EPS 1e-6f

// ---------------------------------------------------------------------------
// helpers
// ---------------------------------------------------------------------------
__device__ __forceinline__ ushort_t f2bf(float f) {
    uint_t u = __float_as_uint(f);
    uint_t r = u + 0x7FFFu + ((u >> 16) & 1u);   // round-to-nearest-even
    return (ushort_t)(r >> 16);
}

__device__ __forceinline__ void atomicMaxF(float* addr, float v) {
    // signed-int max for v>=0, unsigned-int min for v<0; init must be -inf bits
    if (v >= 0.0f) atomicMax((int*)addr, __float_as_int(v));
    else           atomicMin((uint_t*)addr, __float_as_uint(v));
}

union FragU { uint4 q[2]; v16bf v; };

__device__ __forceinline__ v16bf ld_frag(const ushort_t* p, int off2) {
    FragU u;
    u.q[0] = *(const uint4*)(p);
    u.q[1] = *(const uint4*)(p + off2);
    return u.v;
}

// ---------------------------------------------------------------------------
// fill
// ---------------------------------------------------------------------------
__global__ void k_fill(uint_t* p, long n, uint_t v) {
    long t = (long)blockIdx.x * blockDim.x + threadIdx.x;
    if (t < n) p[t] = v;
}

// ---------------------------------------------------------------------------
// weight prep: W[K][N] f32 row-major -> Bt[N][Kpad] bf16 (zero pad k>=K)
// ---------------------------------------------------------------------------
__global__ void k_w2bt(const float* __restrict__ W, ushort_t* __restrict__ Bt,
                       int K, int N, int Kpad) {
    long t = (long)blockIdx.x * blockDim.x + threadIdx.x;
    if (t >= (long)N * Kpad) return;
    int n = (int)(t / Kpad), k = (int)(t % Kpad);
    float v = (k < K) ? W[(long)k * N + n] : 0.0f;
    Bt[t] = f2bf(v);
}

// ---------------------------------------------------------------------------
// WMMA bf16 GEMM:  C[M][ldc] = A[M][lda] @ Bt^T (+bias) , tile 128x64, K-step 32
//   MODE 0: C = acc + bias
//   MODE 1: C = silu(accG) * accU       (two B matrices, no bias)
//   MODE 2: C = (acc + bias) * emul[row*N+col]   (A rows gathered via gidx)
// ---------------------------------------------------------------------------
#define BM 128
#define BN 64
#define BK 32
#define LDS_S 40   // padded row stride (ushorts): 80B -> conflict-free b128 reads

template<int MODE, bool GATHER, bool BIAS>
__global__ __launch_bounds__(256) void k_gemm(
    const float* __restrict__ A, const int* __restrict__ gidx,
    const ushort_t* __restrict__ Bt0, const ushort_t* __restrict__ Bt1,
    const float* __restrict__ bias, const float* __restrict__ emul,
    float* __restrict__ C,
    int M, int N, int K, int lda, int ldc, int Kpad)
{
    constexpr int NB = (MODE == 1) ? 2 : 1;
    __shared__ ushort_t As[BM * LDS_S];
    __shared__ ushort_t Bs[2][BN * LDS_S];

    const int tid  = threadIdx.x;
    const int lane = tid & 31;
    const int wv   = tid >> 5;
    const int mw   = wv >> 1;      // 0..3
    const int nw   = wv & 1;       // 0..1
    const long mBase = (long)blockIdx.y * BM;
    const long nBase = (long)blockIdx.x * BN;

    v8f acc[NB][2][2];
#pragma unroll
    for (int b = 0; b < NB; b++)
#pragma unroll
        for (int i = 0; i < 2; i++)
#pragma unroll
            for (int j = 0; j < 2; j++)
                acc[b][i][j] = (v8f){0.f,0.f,0.f,0.f,0.f,0.f,0.f,0.f};

    // A staging coords: thread -> (row, 16-wide half)
    const int  sRow  = tid >> 1;
    const int  sHalf = (tid & 1) * 16;
    const long aRow  = mBase + sRow;
    const bool aValid = (aRow < M);
    const long aIdx  = aValid ? (GATHER ? (long)gidx[aRow] : aRow) : 0;

    // B staging coords: thread -> (n-row, 8-ushort segment)
    const int  bRow  = tid >> 2;
    const int  bSeg  = (tid & 3) * 8;
    const long gn    = nBase + bRow;
    const bool bValid = (gn < N);

    for (int kk = 0; kk < K; kk += BK) {
        // ---- stage A (f32 -> bf16) ----
        {
            const float* src = A + aIdx * (long)lda + kk + sHalf;
            float v[16];
            if (aValid && ((lda & 3) == 0) && (kk + sHalf + 16 <= K)) {
                const float4* s4 = (const float4*)src;
                float4 a0 = s4[0], a1 = s4[1], a2 = s4[2], a3 = s4[3];
                v[0]=a0.x; v[1]=a0.y; v[2]=a0.z; v[3]=a0.w;
                v[4]=a1.x; v[5]=a1.y; v[6]=a1.z; v[7]=a1.w;
                v[8]=a2.x; v[9]=a2.y; v[10]=a2.z; v[11]=a2.w;
                v[12]=a3.x; v[13]=a3.y; v[14]=a3.z; v[15]=a3.w;
            } else {
#pragma unroll
                for (int e = 0; e < 16; e++)
                    v[e] = (aValid && (kk + sHalf + e) < K) ? src[e] : 0.0f;
            }
            uint_t pk[8];
#pragma unroll
            for (int i = 0; i < 8; i++)
                pk[i] = (uint_t)f2bf(v[2*i]) | ((uint_t)f2bf(v[2*i+1]) << 16);
            uint4 q0, q1;
            q0.x=pk[0]; q0.y=pk[1]; q0.z=pk[2]; q0.w=pk[3];
            q1.x=pk[4]; q1.y=pk[5]; q1.z=pk[6]; q1.w=pk[7];
            uint4* dst = (uint4*)&As[sRow * LDS_S + sHalf];
            dst[0] = q0; dst[1] = q1;
        }
        // ---- stage B (already bf16, transposed, K zero-padded) ----
#pragma unroll
        for (int b = 0; b < NB; b++) {
            const ushort_t* bsrc = (b ? Bt1 : Bt0) + gn * (long)Kpad + kk + bSeg;
            uint4 val;
            if (bValid) val = *(const uint4*)bsrc;
            else { val.x = val.y = val.z = val.w = 0u; }
            *(uint4*)&Bs[b][bRow * LDS_S + bSeg] = val;
        }
        __syncthreads();

        // ---- fragments + WMMA ----
        const int akb = (lane < 16) ? 0 : 8;
        const int bkb = (lane < 16) ? 0 : 8;
        v16bf af[2];
#pragma unroll
        for (int ms = 0; ms < 2; ms++) {
            int m = mw * 32 + ms * 16 + (lane & 15);
            af[ms] = ld_frag(&As[m * LDS_S + akb], 16);
        }
#pragma unroll
        for (int ns = 0; ns < 2; ns++) {
            int n = nw * 32 + ns * 16 + (lane & 15);
#pragma unroll
            for (int b = 0; b < NB; b++) {
                v16bf bfg = ld_frag(&Bs[b][n * LDS_S + bkb], 16);
#pragma unroll
                for (int ms = 0; ms < 2; ms++)
                    acc[b][ms][ns] = __builtin_amdgcn_wmma_f32_16x16x32_bf16(
                        false, af[ms], false, bfg, (short)0, acc[b][ms][ns],
                        false, false);
            }
        }
        __syncthreads();
    }

    // ---- epilogue ----
    const int colOff  = lane & 15;
    const int rowHalf = (lane < 16) ? 0 : 8;
#pragma unroll
    for (int ms = 0; ms < 2; ms++) {
#pragma unroll
        for (int ns = 0; ns < 2; ns++) {
            long col = nBase + nw * 32 + ns * 16 + colOff;
            if (col >= N) continue;
            float bv = BIAS ? bias[col] : 0.0f;
#pragma unroll
            for (int r = 0; r < 8; r++) {
                long row = mBase + mw * 32 + ms * 16 + rowHalf + r;
                if (row >= M) continue;
                float out;
                if (MODE == 1) {
                    float g = acc[0][ms][ns][r];
                    float u = acc[1][ms][ns][r];
                    out = (g / (1.0f + expf(-g))) * u;   // silu(g)*u
                } else {
                    out = acc[0][ms][ns][r] + bv;
                    if (MODE == 2) out *= emul[row * (long)N + col];
                }
                C[row * (long)ldc + col] = out;
            }
        }
    }
}

// ---------------------------------------------------------------------------
// per-row 8-head logit dots: out[row][h] = X[row,:256] . W[:,h]  (W stride 8)
// block = 256 threads = 8 waves, wave w computes head w for one row
// ---------------------------------------------------------------------------
__global__ __launch_bounds__(256) void k_headdots(
    const float* __restrict__ X, const float* __restrict__ W,
    float* __restrict__ out, int rows)
{
    int row = blockIdx.x;
    if (row >= rows) return;
    int lane = threadIdx.x & 31, h = threadIdx.x >> 5;
    const float* x = X + (long)row * DEC;
    float s = 0.0f;
#pragma unroll
    for (int k = lane; k < DEC; k += 32) s += x[k] * W[k * NH + h];
#pragma unroll
    for (int o = 16; o >= 1; o >>= 1) s += __shfl_xor(s, o, 32);
    if (lane == 0) out[(long)row * NH + h] = s;
}

// ---------------------------------------------------------------------------
// HGAT phase A (feedback attention)
// ---------------------------------------------------------------------------
__global__ void kA_self_score(const float* __restrict__ rt_attn,
                              const float* __restrict__ b_attn,
                              float* __restrict__ s_self,
                              float* __restrict__ segmax, long cnt) {
    long t = (long)blockIdx.x * blockDim.x + threadIdx.x;
    if (t >= cnt) return;
    int h = (int)(t & 7);
    float v = rt_attn[t] + b_attn[h];
    v = (v > 0.0f) ? v : 0.2f * v;            // leaky_relu(0.2)
    s_self[t] = v;
    atomicMaxF(&segmax[t], v);
}

__global__ void kA_edge_score(const int* __restrict__ ej, const int* __restrict__ ei,
                              const float* __restrict__ fb_attn,
                              const float* __restrict__ rt_attn,
                              const float* __restrict__ b_attn,
                              float* __restrict__ sA, float* __restrict__ segmax,
                              int E) {
    int e = blockIdx.x * blockDim.x + threadIdx.x;
    if (e >= E) return;
    int j = ej[e], i = ei[e];
#pragma unroll
    for (int h = 0; h < NH; h++) {
        float v = fb_attn[(long)j * NH + h] + rt_attn[(long)i * NH + h] + b_attn[h];
        v = (v > 0.0f) ? v : 0.2f * v;
        sA[(long)e * NH + h] = v;
        atomicMaxF(&segmax[(long)i * NH + h], v);
    }
}

__global__ void k_exp_self(float* __restrict__ s, const float* __restrict__ segmax,
                           float* __restrict__ segsum, long cnt) {
    long t = (long)blockIdx.x * blockDim.x + threadIdx.x;
    if (t >= cnt) return;
    float ex = expf(s[t] - segmax[t]);
    s[t] = ex;
    atomicAdd(&segsum[t], ex);
}

__global__ void k_exp_edge(float* __restrict__ s, const int* __restrict__ ei,
                           const float* __restrict__ segmax,
                           float* __restrict__ segsum, long cnt) {
    long t = (long)blockIdx.x * blockDim.x + threadIdx.x;
    if (t >= cnt) return;
    long e = t >> 3; int h = (int)(t & 7);
    int i = ei[e];
    float ex = expf(s[t] - segmax[(long)i * NH + h]);
    s[t] = ex;
    atomicAdd(&segsum[(long)i * NH + h], ex);
}

// init agg with self-loop contribution (plain store -> also zero-inits buffer)
__global__ __launch_bounds__(256) void kA_self_agg(
    const float* __restrict__ root, const float* __restrict__ s_self,
    const float* __restrict__ segsum, float* __restrict__ agg) {
    int i = blockIdx.x;
#pragma unroll
    for (int c = threadIdx.x; c < ENC; c += 256) {
        int h = c >> 6;
        float w = s_self[(long)i * NH + h] / segsum[(long)i * NH + h];
        agg[(long)i * ENC + c] = w * root[(long)i * ENC + c];
    }
}

__global__ __launch_bounds__(512) void kA_edge_agg(
    const int* __restrict__ ej, const int* __restrict__ ei,
    const float* __restrict__ sA, const float* __restrict__ segsum,
    const float* __restrict__ fbv, float* __restrict__ agg) {
    int e = blockIdx.x, t = threadIdx.x;
    int i = ei[e], j = ej[e];
    int h = t >> 6;
    float w = sA[(long)e * NH + h] / segsum[(long)i * NH + h];
    atomicAdd(&agg[(long)i * ENC + t], w * fbv[(long)j * ENC + t]);
}

// ---------------------------------------------------------------------------
// Self-MHA phase B (edge attention over root graph), qkv layout [n][h*192+d*3+c]
// ---------------------------------------------------------------------------
__global__ __launch_bounds__(256) void kB_score(
    const int* __restrict__ sj, const int* __restrict__ si,
    const float* __restrict__ qkv, const float* __restrict__ eattr,
    float* __restrict__ sB, float* __restrict__ segmax, int E) {
    int e = blockIdx.x * 8 + (threadIdx.x >> 5);
    if (e >= E) return;
    int lane = threadIdx.x & 31;
    int a = si[e], b = sj[e];
    const float* qr = qkv + (long)a * 1536;
    const float* kr = qkv + (long)b * 1536;
    const float* ea = eattr + (long)e * HD;
#pragma unroll
    for (int h = 0; h < NH; h++) {
        float p = 0.0f;
#pragma unroll
        for (int d = lane; d < HD; d += 32)
            p += qr[h * 192 + d * 3] * kr[h * 192 + d * 3 + 1] * ea[d];
#pragma unroll
        for (int o = 16; o >= 1; o >>= 1) p += __shfl_xor(p, o, 32);
        if (lane == 0) {
            float s = p * 0.125f;                 // 1/sqrt(64)
            sB[(long)e * NH + h] = s;
            atomicMaxF(&segmax[(long)a * NH + h], s);
        }
    }
}

__global__ __launch_bounds__(512) void kB_agg(
    const int* __restrict__ sj, const int* __restrict__ si,
    const float* __restrict__ sB, const float* __restrict__ segsum,
    const float* __restrict__ qkv, float* __restrict__ agg) {
    int e = blockIdx.x, t = threadIdx.x;
    int a = si[e], b = sj[e];
    int h = t >> 6, d = t & 63;
    float w = sB[(long)e * NH + h] / segsum[(long)a * NH + h];
    atomicAdd(&agg[(long)a * ENC + t],
              w * qkv[(long)b * 1536 + h * 192 + d * 3 + 2]);
}

// ---------------------------------------------------------------------------
// out[row] = rmsnorm(x[row] + r[row], g)   (C = 512)
// ---------------------------------------------------------------------------
__global__ __launch_bounds__(256) void k_rms(
    const float* __restrict__ x, const float* __restrict__ r,
    const float* __restrict__ g, float* __restrict__ out) {
    long row = blockIdx.x;
    int t = threadIdx.x;
    const float* xr = x + row * ENC;
    const float* rr = r + row * ENC;
    float v0 = xr[t] + rr[t];
    float v1 = xr[t + 256] + rr[t + 256];
    float ss = v0 * v0 + v1 * v1;
#pragma unroll
    for (int o = 16; o >= 1; o >>= 1) ss += __shfl_xor(ss, o, 32);
    __shared__ float part[8];
    int lane = t & 31, w = t >> 5;
    if (lane == 0) part[w] = ss;
    __syncthreads();
    if (t == 0) {
        float tot = 0.0f;
#pragma unroll
        for (int k = 0; k < 8; k++) tot += part[k];
        part[0] = rsqrtf(tot * (1.0f / ENC) + EPS);
    }
    __syncthreads();
    float s = part[0];
    out[row * ENC + t]       = v0 * s * g[t];
    out[row * ENC + t + 256] = v1 * s * g[t + 256];
}

// ---------------------------------------------------------------------------
// launch
// ---------------------------------------------------------------------------
extern "C" void kernel_launch(void* const* d_in, const int* in_sizes, int n_in,
                              void* d_out, int out_size, void* d_ws, size_t ws_size,
                              hipStream_t stream) {
    (void)n_in; (void)out_size; (void)ws_size;

    const float* root   = (const float*)d_in[0];
    const float* fb     = (const float*)d_in[1];
    const int*   fbj    = (const int*)d_in[2];
    const float* fmaps  = (const float*)d_in[3];
    const int*   r2f    = (const int*)d_in[4];
    const int*   rej    = (const int*)d_in[5];
    const float* eattr  = (const float*)d_in[6];
    const float* W_c2x  = (const float*)d_in[7];
    const float* b_c2x  = (const float*)d_in[8];
    const float* W_x2c  = (const float*)d_in[9];
    const float* b_x2c  = (const float*)d_in[10];
    const float* W_attn = (const float*)d_in[11];
    const float* b_attn = (const float*)d_in[12];
    const float* W_qkv  = (const float*)d_in[13];
    const float* b_qkv  = (const float*)d_in[14];
    const float* W_gate = (const float*)d_in[15];
    const float* W_up   = (const float*)d_in[16];
    const float* W_out  = (const float*)d_in[17];
    const float* W_fr   = (const float*)d_in[18];
    const float* b_fr   = (const float*)d_in[19];
    const float* g_node = (const float*)d_in[20];
    const float* g_root = (const float*)d_in[21];
    const float* g_ffn  = (const float*)d_in[22];

    const int Nn  = in_sizes[0] / ENC;     // 16384
    const int M   = in_sizes[1] / DEC;     // 65536
    const int E1  = in_sizes[2] / 2;       // 131072
    const int F   = in_sizes[3] / DEC;     // 16384
    const int E2  = in_sizes[5] / 2;       // 131072
    const int HID = in_sizes[15] / ENC;    // 1365
    const int HIDp = (HID + 31) & ~31;     // 1376
    const int* fbi = fbj + E1;             // feedback_index: [j; i]
    const int* rei = rej + E2;             // root_edge_index: [sj; si]

    // ---- scratch bump allocator ----
    size_t off = 0;
    char* base = (char*)d_ws;
    auto alloc = [&](size_t bytes) -> void* {
        off = (off + 255) & ~(size_t)255;
        void* p = base + off;
        off += bytes;
        return p;
    };
    ushort_t* bt_x2c  = (ushort_t*)alloc((size_t)DEC * ENC * 2);
    ushort_t* bt_c2x  = (ushort_t*)alloc((size_t)ENC * DEC * 2);
    ushort_t* bt_qkv  = (ushort_t*)alloc((size_t)(3 * ENC) * ENC * 2);
    ushort_t* bt_gate = (ushort_t*)alloc((size_t)HID * ENC * 2);
    ushort_t* bt_up   = (ushort_t*)alloc((size_t)HID * ENC * 2);
    ushort_t* bt_out  = (ushort_t*)alloc((size_t)ENC * HIDp * 2);
    ushort_t* bt_fr   = (ushort_t*)alloc((size_t)DEC * ENC * 2);
    float* fb_attn = (float*)alloc((size_t)M * NH * 4);
    float* rt_attn = (float*)alloc((size_t)Nn * NH * 4);
    float* sA      = (float*)alloc((size_t)E1 * NH * 4);
    float* s_self  = (float*)alloc((size_t)Nn * NH * 4);
    float* segmaxA = (float*)alloc((size_t)Nn * NH * 4);
    float* segsumA = (float*)alloc((size_t)Nn * NH * 4);
    float* segmaxB = (float*)alloc((size_t)Nn * NH * 4);
    float* segsumB = (float*)alloc((size_t)Nn * NH * 4);
    float* sB      = (float*)alloc((size_t)E2 * NH * 4);
    float* ctx_root= (float*)alloc((size_t)Nn * DEC * 4);
    float* x1      = (float*)alloc((size_t)Nn * ENC * 4);
    float* x2      = (float*)alloc((size_t)Nn * ENC * 4);
    float* aggbuf  = (float*)alloc((size_t)Nn * ENC * 4);   // agg -> agg2 -> y
    size_t bigN = (size_t)M * ENC;                           // fb_c2x (largest)
    if ((size_t)Nn * 3 * ENC > bigN) bigN = (size_t)Nn * 3 * ENC;
    if ((size_t)Nn * HIDp    > bigN) bigN = (size_t)Nn * HIDp;
    float* big = (float*)alloc(bigN * 4);                    // fb_c2x / qkv / h

    float* x3     = (float*)d_out;                 // [Nn,512]
    float* fringe = x3 + (size_t)Nn * ENC;         // [F,256]

    dim3 B256(256), B512(512);
    auto cdiv = [](long a, long b) { return (unsigned)((a + b - 1) / b); };
    auto ggrid = [&](int m, int n) { return dim3(cdiv(n, BN), cdiv(m, BM)); };

    // ---- weight prep (f32 -> bf16, transposed, K zero-padded) ----
    k_w2bt<<<cdiv((long)DEC * ENC, 256), B256, 0, stream>>>(W_x2c, bt_x2c, ENC, DEC, ENC);
    k_w2bt<<<cdiv((long)ENC * DEC, 256), B256, 0, stream>>>(W_c2x, bt_c2x, DEC, ENC, DEC);
    k_w2bt<<<cdiv((long)3 * ENC * ENC, 256), B256, 0, stream>>>(W_qkv, bt_qkv, ENC, 3 * ENC, ENC);
    k_w2bt<<<cdiv((long)HID * ENC, 256), B256, 0, stream>>>(W_gate, bt_gate, ENC, HID, ENC);
    k_w2bt<<<cdiv((long)HID * ENC, 256), B256, 0, stream>>>(W_up, bt_up, ENC, HID, ENC);
    k_w2bt<<<cdiv((long)ENC * HIDp, 256), B256, 0, stream>>>(W_out, bt_out, HID, ENC, HIDp);
    k_w2bt<<<cdiv((long)DEC * ENC, 256), B256, 0, stream>>>(W_fr, bt_fr, ENC, DEC, ENC);

    // ---- phase A: HGAT feedback attention ----
    // ctx_root = root @ W_x2c + b_x2c
    k_gemm<0, false, true><<<ggrid(Nn, DEC), B256, 0, stream>>>(
        root, nullptr, bt_x2c, nullptr, b_x2c, nullptr, ctx_root,
        Nn, DEC, ENC, ENC, DEC, ENC);
    // fb_c2x = feedback @ W_c2x + b_c2x   (values, per feedback node)
    k_gemm<0, false, true><<<ggrid(M, ENC), B256, 0, stream>>>(
        fb, nullptr, bt_c2x, nullptr, b_c2x, nullptr, big,
        M, ENC, DEC, DEC, ENC, DEC);
    // attention logit halves
    k_headdots<<<M, B256, 0, stream>>>(fb, W_attn, fb_attn, M);
    k_headdots<<<Nn, B256, 0, stream>>>(ctx_root, W_attn + (size_t)DEC * NH, rt_attn, Nn);
    // segment max/sum init
    k_fill<<<cdiv((long)Nn * NH, 256), B256, 0, stream>>>((uint_t*)segmaxA, (long)Nn * NH, 0xFF800000u);
    k_fill<<<cdiv((long)Nn * NH, 256), B256, 0, stream>>>((uint_t*)segsumA, (long)Nn * NH, 0u);
    // scores + segment max
    kA_self_score<<<cdiv((long)Nn * NH, 256), B256, 0, stream>>>(rt_attn, b_attn, s_self, segmaxA, (long)Nn * NH);
    kA_edge_score<<<cdiv(E1, 256), B256, 0, stream>>>(fbj, fbi, fb_attn, rt_attn, b_attn, sA, segmaxA, E1);
    // exp + segment sum
    k_exp_self<<<cdiv((long)Nn * NH, 256), B256, 0, stream>>>(s_self, segmaxA, segsumA, (long)Nn * NH);
    k_exp_edge<<<cdiv((long)E1 * NH, 256), B256, 0, stream>>>(sA, fbi, segmaxA, segsumA, (long)E1 * NH);
    // aggregate (self init + edge scatter-add)
    kA_self_agg<<<Nn, B256, 0, stream>>>(root, s_self, segsumA, aggbuf);
    kA_edge_agg<<<E1, B512, 0, stream>>>(fbj, fbi, sA, segsumA, big, aggbuf);
    // x1 = rmsnorm(root + agg, g_node)
    k_rms<<<Nn, B256, 0, stream>>>(root, aggbuf, g_node, x1);

    // ---- phase B: self-MHA over root graph ----
    // qkv = x1 @ W_qkv + b_qkv   (reuses `big`)
    k_gemm<0, false, true><<<ggrid(Nn, 3 * ENC), B256, 0, stream>>>(
        x1, nullptr, bt_qkv, nullptr, b_qkv, nullptr, big,
        Nn, 3 * ENC, ENC, ENC, 3 * ENC, ENC);
    k_fill<<<cdiv((long)Nn * NH, 256), B256, 0, stream>>>((uint_t*)segmaxB, (long)Nn * NH, 0xFF800000u);
    k_fill<<<cdiv((long)Nn * NH, 256), B256, 0, stream>>>((uint_t*)segsumB, (long)Nn * NH, 0u);
    k_fill<<<cdiv((long)Nn * ENC, 256), B256, 0, stream>>>((uint_t*)aggbuf, (long)Nn * ENC, 0u);
    kB_score<<<cdiv(E2, 8), B256, 0, stream>>>(rej, rei, big, eattr, sB, segmaxB, E2);
    k_exp_edge<<<cdiv((long)E2 * NH, 256), B256, 0, stream>>>(sB, rei, segmaxB, segsumB, (long)E2 * NH);
    kB_agg<<<E2, B512, 0, stream>>>(rej, rei, sB, segsumB, big, aggbuf);
    // x2 = rmsnorm(x1 + agg2, g_root)
    k_rms<<<Nn, B256, 0, stream>>>(x1, aggbuf, g_root, x2);

    // ---- phase C: SwiGLU FFN ----
    // h = silu(x2 @ W_gate) * (x2 @ W_up)   (fused, ldc = HIDp, reuses `big`)
    k_gemm<1, false, false><<<ggrid(Nn, HID), B256, 0, stream>>>(
        x2, nullptr, bt_gate, bt_up, nullptr, nullptr, big,
        Nn, HID, ENC, ENC, HIDp, ENC);
    // y = h @ W_out   (ragged K=1365, Kpad=1376; reuses aggbuf)
    k_gemm<0, false, false><<<ggrid(Nn, ENC), B256, 0, stream>>>(
        big, nullptr, bt_out, nullptr, nullptr, nullptr, aggbuf,
        Nn, ENC, HID, HIDp, ENC, HIDp);
    // x3 = rmsnorm(x2 + y, g_ffn)  -> d_out
    k_rms<<<Nn, B256, 0, stream>>>(x2, aggbuf, g_ffn, x3);

    // ---- phase D: fringe = (x3[r2f] @ W_fr + b_fr) * fringe_maps ----
    k_gemm<2, true, true><<<ggrid(F, DEC), B256, 0, stream>>>(
        x3, r2f, bt_fr, nullptr, b_fr, fmaps, fringe,
        F, DEC, ENC, ENC, DEC, ENC);
}
